// GCNNet_34626026340852
// MI455X (gfx1250) — matrix operand order repaired and checked by reading, not compiled, not run on hardware
//
#include <hip/hip_runtime.h>
#include <hip/hip_bf16.h>

typedef __attribute__((ext_vector_type(16))) _Float16 v16h;
typedef __attribute__((ext_vector_type(8)))  _Float16 v8h;
typedef __attribute__((ext_vector_type(8)))  float    v8f;

#define WMMA_F16(a, b, c) \
  __builtin_amdgcn_wmma_f32_16x16x32_f16(false, (a), false, (b), (short)0, (c), false, false)

// ---------------------------------------------------------------------------
// WMMA fragment loaders from pre-converted f16 buffers. No bounds checks:
// all operands are zero-padded to Kp=ceil32(K), Np=ceil16(N) at prep time.
// K order contiguous: lane half h holds k = 16h..16h+15 (A and B).
// ---------------------------------------------------------------------------

// A: 16x32 (MxK) tile, row-major f16, lda % 8 == 0 -> two b128 loads.
__device__ __forceinline__ v16h load_a_h(const _Float16* __restrict__ A, int lda, int k0) {
  const int lane = threadIdx.x & 31;
  const int row  = lane & 15;
  const int kb   = (lane >> 4) * 16;
  const _Float16* p = A + (long long)row * lda + k0 + kb;
  v8h lo = *(const v8h*)p;         // K = kb .. kb+7
  v8h hi = *(const v8h*)(p + 8);   // K = kb+8 .. kb+15
  v16h a;
#pragma unroll
  for (int j = 0; j < 8; ++j) { a[j] = lo[j]; a[8 + j] = hi[j]; }
  return a;
}

// B: 32x16 (KxN) tile from TRANSPOSED f16 weights Wt[N][Kp] -> contiguous
// 16 halves per lane = two b128 loads.  ldt % 8 == 0.
__device__ __forceinline__ v16h load_bT_h(const _Float16* __restrict__ Wt, int ldt,
                                          int k0, int n0) {
  const int lane = threadIdx.x & 31;
  const int n    = n0 + (lane & 15);
  const int kb   = (lane >> 4) * 16;
  const _Float16* p = Wt + (long long)n * ldt + k0 + kb;
  v8h lo = *(const v8h*)p;
  v8h hi = *(const v8h*)(p + 8);
  v16h b;
#pragma unroll
  for (int j = 0; j < 8; ++j) { b[j] = lo[j]; b[8 + j] = hi[j]; }
  return b;
}

// ---------------------------------------------------------------------------
// GEMM: C = act(A[M,Kp](f16) @ Wt[Np,Kp]^T(f16) + bias). One wave: 64x16 tile.
// grid: (Np/16, M/64), block 32. OUTF16: store f16 activation, else f32.
// ---------------------------------------------------------------------------
template <bool OUTF16>
__global__ void __launch_bounds__(32)
gemm_wmma(const _Float16* __restrict__ A, int lda,
          const _Float16* __restrict__ Wt, int ldt,
          const float* __restrict__ bias,
          void* __restrict__ Cv, int ldc,
          int M, int N, int Kp, int relu) {
  const int n0 = blockIdx.x * 16;
  const int m0 = blockIdx.y * 64;
  v8f acc[4] = {};
  for (int k0 = 0; k0 < Kp; k0 += 32) {
    v16h b = load_bT_h(Wt, ldt, k0, n0);
#pragma unroll
    for (int t = 0; t < 4; ++t) {
      v16h a = load_a_h(A + (long long)(m0 + 16 * t) * lda, lda, k0);
      acc[t] = WMMA_F16(a, b, acc[t]);
    }
  }
  const int lane = threadIdx.x & 31;
  const int n  = n0 + (lane & 15);
  const int mb = m0 + (lane >> 4) * 8;
  if (n < N) {
    const float bv = bias[n];
#pragma unroll
    for (int t = 0; t < 4; ++t) {
#pragma unroll
      for (int v = 0; v < 8; ++v) {
        float xv = acc[t][v] + bv;
        if (relu) xv = fmaxf(xv, 0.0f);
        long long off = (long long)(mb + 16 * t + v) * ldc + n;
        if (OUTF16) ((_Float16*)Cv)[off] = (_Float16)xv;
        else        ((float*)Cv)[off] = xv;
      }
    }
  }
}

// ---------------------------------------------------------------------------
// Conv1d (VALID, K=8) implicit GEMM. f16 input slab in LDS; one wave computes
// 32 out-channels x 16 positions. Weights pre-converted f16 [Cout][Kp], K-padded.
// grid: (ceil(Lout/16), Cout/32, B), block 32.  TRANS: input is [B, Lin, CIN].
// ---------------------------------------------------------------------------
template <int CIN, bool TRANS>
__global__ void __launch_bounds__(32)
conv_wmma(const float* __restrict__ in, const _Float16* __restrict__ wt,
          const float* __restrict__ bias, float* __restrict__ out,
          int Lin, int Lout, int Cout) {
  constexpr int Kred = CIN * 8;
  constexpr int Kp   = (Kred + 31) & ~31;
  const int b  = blockIdx.z;
  const int o0 = blockIdx.y * 32;
  const int p0 = blockIdx.x * 16;
  const int lane = threadIdx.x;

  __shared__ _Float16 sIn[CIN][24];   // positions p0..p0+22, all input channels
  for (int idx = lane; idx < CIN * 23; idx += 32) {
    int i = idx / 23, j = idx % 23;
    int p = p0 + j;
    float v = 0.0f;
    if (p < Lin)
      v = TRANS ? in[((long long)b * Lin + p) * CIN + i]
                : in[((long long)b * CIN + i) * Lin + p];
    sIn[i][j] = (_Float16)v;
  }
  __syncthreads();

  const int n    = lane & 15;
  const int kb16 = (lane >> 4) * 16;
  v8f acc0 = {}, acc1 = {};
  for (int k0 = 0; k0 < Kp; k0 += 32) {
    v16h bf;
#pragma unroll
    for (int j = 0; j < 16; ++j) {
      int k = k0 + kb16 + j;                        // k = i*8 + t
      if (Kp != Kred) k = (k < Kred) ? k : Kred - 1; // pad region: weights are 0
      bf[j] = sIn[k >> 3][(k & 7) + n];
    }
    v16h a0 = load_a_h(wt + (long long)o0 * Kp, Kp, k0);
    acc0 = WMMA_F16(a0, bf, acc0);
    v16h a1 = load_a_h(wt + (long long)(o0 + 16) * Kp, Kp, k0);
    acc1 = WMMA_F16(a1, bf, acc1);
  }

  const int p  = p0 + n;
  const int ob = o0 + (lane >> 4) * 8;
  if (p < Lout) {
#pragma unroll
    for (int v = 0; v < 8; ++v) {
      int oA = ob + v, oB = ob + 16 + v;
      out[((long long)b * Cout + oA) * Lout + p] = fmaxf(acc0[v] + bias[oA], 0.0f);
      out[((long long)b * Cout + oB) * Lout + p] = fmaxf(acc1[v] + bias[oB], 0.0f);
    }
  }
}

// ---------------------------------------------------------------------------
// Weight / activation prep kernels (run once per call; operands tiny)
// ---------------------------------------------------------------------------
// Wt[n*Kp + k] = (k<K && n<N) ? W[k*N + n] : 0   (f32 [K,N] -> f16 [Np,Kp] T)
__global__ void k_wprep(const float* __restrict__ W, _Float16* __restrict__ Wt,
                        int K, int N, int Kp, int Np) {
  long long idx = (long long)blockIdx.x * blockDim.x + threadIdx.x;
  if (idx >= (long long)Np * Kp) return;
  int n = (int)(idx / Kp), k = (int)(idx % Kp);
  Wt[idx] = (k < K && n < N) ? (_Float16)W[(long long)k * N + n] : (_Float16)0.0f;
}
// dst[r*Kp + k] = k<K ? src[r*K + k] : 0   (row-major f32 -> f16, K-padded)
__global__ void k_rowpad(const float* __restrict__ src, _Float16* __restrict__ dst,
                         int rows, int K, int Kp) {
  long long idx = (long long)blockIdx.x * blockDim.x + threadIdx.x;
  if (idx >= (long long)rows * Kp) return;
  int r = (int)(idx / Kp), k = (int)(idx % Kp);
  dst[idx] = (k < K) ? (_Float16)src[(long long)r * K + k] : (_Float16)0.0f;
}

// ---------------------------------------------------------------------------
// Elementwise / graph kernels
// ---------------------------------------------------------------------------
__global__ void k_zero_f(float* p, long long n) {
  long long i = (long long)blockIdx.x * blockDim.x + threadIdx.x;
  if (i < n) p[i] = 0.0f;
}
__global__ void k_zero_u(unsigned* p, long long n) {
  long long i = (long long)blockIdx.x * blockDim.x + threadIdx.x;
  if (i < n) p[i] = 0u;
}
__global__ void k_deg_count(const int* __restrict__ dst, unsigned* __restrict__ deg, int E) {
  int e = blockIdx.x * blockDim.x + threadIdx.x;
  if (e < E) atomicAdd(&deg[dst[e]], 1u);
}
__global__ void k_dis(const unsigned* __restrict__ deg, float* __restrict__ dis, int N) {
  int i = blockIdx.x * blockDim.x + threadIdx.x;
  if (i < N) dis[i] = rsqrtf((float)deg[i] + 1.0f);   // +1 self loop
}

template <int FIN, int FOUT>
__global__ void k_xw(const float* __restrict__ x, const float* __restrict__ W,
                     float* __restrict__ h, int N) {
  int node = blockIdx.x * blockDim.x + threadIdx.x;
  if (node >= N) return;
  float xi[FIN];
#pragma unroll
  for (int f = 0; f < FIN; ++f) xi[f] = x[(long long)node * FIN + f];
#pragma unroll
  for (int o = 0; o < FOUT; ++o) {
    float s = 0.0f;
#pragma unroll
    for (int f = 0; f < FIN; ++f) s += xi[f] * W[f * FOUT + o];
    h[(long long)node * FOUT + o] = s;
  }
}

template <int FOUT>
__global__ void k_scatter(const int* __restrict__ src, const int* __restrict__ dst,
                          const float* __restrict__ dis, const float* __restrict__ h,
                          float* __restrict__ agg, int E) {
  int e = blockIdx.x * blockDim.x + threadIdx.x;
  if (e >= E) return;
  int s = src[e], d = dst[e];
  float c = dis[s] * dis[d];
  const float* hs = h + (long long)s * FOUT;
  float* ad = agg + (long long)d * FOUT;
#pragma unroll
  for (int f = 0; f < FOUT; ++f)
    __hip_atomic_fetch_add(&ad[f], c * hs[f], __ATOMIC_RELAXED, __HIP_MEMORY_SCOPE_AGENT);
}

template <int FOUT>
__global__ void k_finalize(const float* __restrict__ agg, const float* __restrict__ h,
                           const float* __restrict__ dis, const float* __restrict__ bias,
                           float* __restrict__ out, int N) {
  int node = blockIdx.x * blockDim.x + threadIdx.x;
  if (node >= N) return;
  float d2 = dis[node] * dis[node];
#pragma unroll
  for (int f = 0; f < FOUT; ++f) {
    long long i = (long long)node * FOUT + f;
    out[i] = fmaxf(agg[i] + d2 * h[i] + bias[f], 0.0f);
  }
}

// global_max_pool over batch ids; values are post-ReLU (>=0) so uint-bit max works
__global__ void k_gmax(const float* __restrict__ h, const int* __restrict__ batch,
                       unsigned* __restrict__ g, int N) {
  int node = blockIdx.x * blockDim.x + threadIdx.x;
  if (node >= N) return;
  int b = batch[node];
#pragma unroll
  for (int f = 0; f < 16; ++f)
    atomicMax(&g[b * 16 + f], __float_as_uint(h[(long long)node * 16 + f]));
}

// MaxPool1d(3, stride 3) over last dim; rows = B*C
__global__ void k_mp3(const float* __restrict__ in, float* __restrict__ out,
                      long long rows, int Lin, int Lout) {
  long long idx = (long long)blockIdx.x * blockDim.x + threadIdx.x;
  if (idx >= rows * Lout) return;
  long long r = idx / Lout;
  int p = (int)(idx % Lout);
  const float* x = in + r * Lin + p * 3;
  out[idx] = fmaxf(fmaxf(x[0], x[1]), x[2]);
}

// mean over last dim, f16 output (feeds GEMM A operand)
__global__ void k_mean_h(const float* __restrict__ in, _Float16* __restrict__ out,
                         long long rows, int Lp) {
  long long idx = (long long)blockIdx.x * blockDim.x + threadIdx.x;
  if (idx >= rows) return;
  const float* x = in + idx * Lp;
  float s = 0.0f;
  for (int i = 0; i < Lp; ++i) s += x[i];
  out[idx] = (_Float16)(s / (float)Lp);
}

// ---------------------------------------------------------------------------
// Host: orchestration
// ---------------------------------------------------------------------------
static inline int ceilDiv(int a, int b) { return (a + b - 1) / b; }
static inline char* wsTake(char*& p, size_t bytes) {
  char* r = p;
  p += (bytes + 255) & ~size_t(255);
  return r;
}

extern "C" void kernel_launch(void* const* d_in, const int* in_sizes, int n_in,
                              void* d_out, int out_size, void* d_ws, size_t ws_size,
                              hipStream_t stream) {
  (void)in_sizes; (void)n_in; (void)out_size; (void)ws_size;
  constexpr int N = 200000, E = 3200000, B = 512, L = 1000;

  const float* x      = (const float*)d_in[0];
  const int*   eidx   = (const int*)d_in[1];
  const int*   src    = eidx;
  const int*   dst    = eidx + E;
  const int*   batch  = (const int*)d_in[2];
  const float* target = (const float*)d_in[3];
  const float* W1 = (const float*)d_in[4];  const float* b1 = (const float*)d_in[5];
  const float* W2 = (const float*)d_in[6];  const float* b2 = (const float*)d_in[7];
  const float* W3 = (const float*)d_in[8];  const float* b3 = (const float*)d_in[9];
  const float* Wg1 = (const float*)d_in[10]; const float* bg1 = (const float*)d_in[11];
  const float* Wg2 = (const float*)d_in[12]; const float* bg2 = (const float*)d_in[13];
  const float* K1 = (const float*)d_in[14]; const float* cb1 = (const float*)d_in[15];
  const float* K2 = (const float*)d_in[16]; const float* cb2 = (const float*)d_in[17];
  const float* K3 = (const float*)d_in[18]; const float* cb3 = (const float*)d_in[19];
  const float* Wxt = (const float*)d_in[20]; const float* bxt = (const float*)d_in[21];
  const float* Wf1 = (const float*)d_in[22]; const float* bf1 = (const float*)d_in[23];
  const float* Wf2 = (const float*)d_in[24]; const float* bf2 = (const float*)d_in[25];
  const float* Wo = (const float*)d_in[26]; const float* bo = (const float*)d_in[27];
  float* out = (float*)d_out;

  // workspace carve-up
  char* wp = (char*)d_ws;
  float*    dis   = (float*)wsTake(wp, (size_t)N * 4);
  unsigned* deg   = (unsigned*)wsTake(wp, (size_t)N * 4);
  float*    hbuf  = (float*)wsTake(wp, (size_t)N * 16 * 4);
  float*    agg   = (float*)wsTake(wp, (size_t)N * 16 * 4);
  float*    nodeA = (float*)wsTake(wp, (size_t)N * 16 * 4);
  float*    nodeB = (float*)wsTake(wp, (size_t)N * 16 * 4);
  unsigned* gbits = (unsigned*)wsTake(wp, (size_t)B * 16 * 4);
  // f16 activations
  _Float16* g16   = (_Float16*)wsTake(wp, (size_t)B * 32 * 2);     // K=16 pad to 32
  _Float16* g1h   = (_Float16*)wsTake(wp, (size_t)B * 1024 * 2);
  _Float16* fusedh= (_Float16*)wsTake(wp, (size_t)B * 256 * 2);
  _Float16* f1h   = (_Float16*)wsTake(wp, (size_t)B * 1024 * 2);
  _Float16* f2h   = (_Float16*)wsTake(wp, (size_t)B * 512 * 2);
  _Float16* cm16  = (_Float16*)wsTake(wp, (size_t)B * 128 * 2);
  // f16 transposed, padded weights  Wt[Np][Kp]
  _Float16* Wg1t  = (_Float16*)wsTake(wp, (size_t)1024 * 32 * 2);
  _Float16* Wg2t  = (_Float16*)wsTake(wp, (size_t)128 * 1024 * 2);
  _Float16* Wxtt  = (_Float16*)wsTake(wp, (size_t)128 * 128 * 2);
  _Float16* Wf1t  = (_Float16*)wsTake(wp, (size_t)1024 * 256 * 2);
  _Float16* Wf2t  = (_Float16*)wsTake(wp, (size_t)512 * 1024 * 2);
  _Float16* Wot   = (_Float16*)wsTake(wp, (size_t)16 * 512 * 2);
  // f16 conv weights [Cout][Kp]
  _Float16* cw1   = (_Float16*)wsTake(wp, (size_t)32 * 64 * 2);
  _Float16* cw2   = (_Float16*)wsTake(wp, (size_t)64 * 256 * 2);
  _Float16* cw3   = (_Float16*)wsTake(wp, (size_t)128 * 512 * 2);
  // conv f32 ping-pong
  float*    convA = (float*)wsTake(wp, (size_t)B * 32 * 993 * 4);
  float*    convB = (float*)wsTake(wp, (size_t)B * 32 * 331 * 4);

  const int T = 256;
  auto gemm16 = [&](const _Float16* A, int lda, const _Float16* Wt, int ldt,
                    const float* bias, _Float16* C, int ldc, int M, int Ncol, int Kp,
                    int relu) {
    dim3 grid(ceilDiv(Ncol, 16), M / 64);
    gemm_wmma<true><<<grid, 32, 0, stream>>>(A, lda, Wt, ldt, bias, C, ldc, M, Ncol, Kp, relu);
  };
  auto wprep = [&](const float* W, _Float16* Wt, int K, int Ncol, int Kp, int Np) {
    k_wprep<<<ceilDiv(Np * Kp, T), T, 0, stream>>>(W, Wt, K, Ncol, Kp, Np);
  };

  // ---- weight prep (f16, transposed, padded) ------------------------------
  wprep(Wg1, Wg1t, 16, 1024, 32, 1024);
  wprep(Wg2, Wg2t, 1024, 128, 1024, 128);
  wprep(Wxt, Wxtt, 128, 128, 128, 128);
  wprep(Wf1, Wf1t, 256, 1024, 256, 1024);
  wprep(Wf2, Wf2t, 1024, 512, 1024, 512);
  wprep(Wo,  Wot,  512, 2,   512, 16);
  k_rowpad<<<ceilDiv(32 * 64, T), T, 0, stream>>>(K1, cw1, 32, 40, 64);
  k_rowpad<<<ceilDiv(64 * 256, T), T, 0, stream>>>(K2, cw2, 64, 256, 256);
  k_rowpad<<<ceilDiv(128 * 512, T), T, 0, stream>>>(K3, cw3, 128, 512, 512);

  // ---- degree / normalization --------------------------------------------
  k_zero_u<<<ceilDiv(N, T), T, 0, stream>>>(deg, N);
  k_deg_count<<<ceilDiv(E, T), T, 0, stream>>>(dst, deg, E);
  k_dis<<<ceilDiv(N, T), T, 0, stream>>>(deg, dis, N);

  // ---- GCN layer 1: 4 -> 4 -----------------------------------------------
  k_xw<4, 4><<<ceilDiv(N, T), T, 0, stream>>>(x, W1, hbuf, N);
  k_zero_f<<<ceilDiv(N * 4, T), T, 0, stream>>>(agg, (long long)N * 4);
  k_scatter<4><<<ceilDiv(E, T), T, 0, stream>>>(src, dst, dis, hbuf, agg, E);
  k_finalize<4><<<ceilDiv(N, T), T, 0, stream>>>(agg, hbuf, dis, b1, nodeA, N);

  // ---- GCN layer 2: 4 -> 8 -----------------------------------------------
  k_xw<4, 8><<<ceilDiv(N, T), T, 0, stream>>>(nodeA, W2, hbuf, N);
  k_zero_f<<<ceilDiv(N * 8, T), T, 0, stream>>>(agg, (long long)N * 8);
  k_scatter<8><<<ceilDiv(E, T), T, 0, stream>>>(src, dst, dis, hbuf, agg, E);
  k_finalize<8><<<ceilDiv(N, T), T, 0, stream>>>(agg, hbuf, dis, b2, nodeB, N);

  // ---- GCN layer 3: 8 -> 16 ----------------------------------------------
  k_xw<8, 16><<<ceilDiv(N, T), T, 0, stream>>>(nodeB, W3, hbuf, N);
  k_zero_f<<<ceilDiv(N * 16, T), T, 0, stream>>>(agg, (long long)N * 16);
  k_scatter<16><<<ceilDiv(E, T), T, 0, stream>>>(src, dst, dis, hbuf, agg, E);
  k_finalize<16><<<ceilDiv(N, T), T, 0, stream>>>(agg, hbuf, dis, b3, nodeA, N);

  // ---- global max pool + graph head --------------------------------------
  k_zero_u<<<ceilDiv(B * 16, T), T, 0, stream>>>(gbits, B * 16);
  k_gmax<<<ceilDiv(N, T), T, 0, stream>>>(nodeA, batch, gbits, N);
  k_rowpad<<<ceilDiv(B * 32, T), T, 0, stream>>>((const float*)gbits, g16, B, 16, 32);
  gemm16(g16, 32, Wg1t, 32, bg1, g1h, 1024, B, 1024, 32, 1);
  gemm16(g1h, 1024, Wg2t, 1024, bg2, fusedh, 256, B, 128, 1024, 0);   // fused[:,0:128]

  // ---- sequence branch ----------------------------------------------------
  {
    dim3 grid(ceilDiv(993, 16), 1, B);     // conv1: [B,L,5] -> [B,32,993]
    conv_wmma<5, true><<<grid, 32, 0, stream>>>(target, cw1, cb1, convA, L, 993, 32);
  }
  k_mp3<<<ceilDiv(B * 32 * 331, T), T, 0, stream>>>(convA, convB, (long long)B * 32, 993, 331);
  {
    dim3 grid(ceilDiv(324, 16), 2, B);     // conv2: [B,32,331] -> [B,64,324]
    conv_wmma<32, false><<<grid, 32, 0, stream>>>(convB, cw2, cb2, convA, 331, 324, 64);
  }
  k_mp3<<<ceilDiv(B * 64 * 108, T), T, 0, stream>>>(convA, convB, (long long)B * 64, 324, 108);
  {
    dim3 grid(ceilDiv(101, 16), 4, B);     // conv3: [B,64,108] -> [B,128,101]
    conv_wmma<64, false><<<grid, 32, 0, stream>>>(convB, cw3, cb3, convA, 108, 101, 128);
  }
  k_mp3<<<ceilDiv(B * 128 * 33, T), T, 0, stream>>>(convA, convB, (long long)B * 128, 101, 33);
  k_mean_h<<<ceilDiv(B * 128, T), T, 0, stream>>>(convB, cm16, (long long)B * 128, 33);
  gemm16(cm16, 128, Wxtt, 128, bxt, fusedh + 128, 256, B, 128, 128, 1); // fused[:,128:256]

  // ---- fusion head --------------------------------------------------------
  gemm16(fusedh, 256, Wf1t, 256, bf1, f1h, 1024, B, 1024, 256, 1);
  gemm16(f1h, 1024, Wf2t, 1024, bf2, f2h, 512, B, 512, 1024, 1);
  {
    dim3 grid(1, B / 64);                  // final: N=2, f32 output
    gemm_wmma<false><<<grid, 32, 0, stream>>>(f2h, 512, Wot, 512, bo, out, 2, B, 2, 512, 0);
  }
}
